// RTMCCBlock_41712722378763
// MI455X (gfx1250) — matrix operand order repaired and checked by reading, not compiled, not run on hardware
//
#include <hip/hip_runtime.h>
#include <hip/hip_bf16.h>
#include <math.h>

typedef __attribute__((ext_vector_type(16))) _Float16 v16h;
typedef __attribute__((ext_vector_type(8)))  _Float16 v8h;
typedef __attribute__((ext_vector_type(8)))  float    v8f;

#define BB 8
#define NN 4096
#define DD 256
#define EE 512
#define SS 128
#define NC (2*EE + SS)   // 1152 output cols of the uv projection
#define VLD 72           // padded LDS leading dim (halves)

static __device__ __forceinline__ v16h frag_cat(v8h lo, v8h hi) {
  union { v16h v; v8h h[2]; } u;
  u.h[0] = lo; u.h[1] = hi;
  return u.v;
}

// A fragment: 16x32 f16 tile, logical row-major [row, k], leading dim ld (halves).
static __device__ __forceinline__ v16h load_frag_a(const _Float16* base, int ld) {
  int l  = threadIdx.x & 31;
  int r  = l & 15;
  int kg = (l >> 4) << 3;                      // 0 or 8
  const _Float16* p = base + (size_t)r * ld + kg;
  v8h lo = *(const v8h*)(p);                   // K = kg .. kg+7
  v8h hi = *(const v8h*)(p + 16);              // K = 16+kg .. 16+kg+7
  return frag_cat(lo, hi);
}

// B fragment: 32x16 (KxN), read from "Bt" stored row-major [col, k] (K contiguous).
static __device__ __forceinline__ v16h load_frag_b(const _Float16* base, int ld) {
  int l  = threadIdx.x & 31;
  int c  = l & 15;
  int kg = (l >> 4) << 3;                      // 0 or 8
  const _Float16* p = base + (size_t)c * ld + kg;
  v8h lo = *(const v8h*)(p);
  v8h hi = *(const v8h*)(p + 16);
  return frag_cat(lo, hi);
}

static __device__ __forceinline__ v8f wmma16(v16h a, v16h b, v8f c) {
  return __builtin_amdgcn_wmma_f32_16x16x32_f16(false, a, false, b, (short)0, c,
                                                false, false);
}

// async global->LDS copy of 16 bytes (per lane), tracked by ASYNCcnt
static __device__ __forceinline__ void async_b128(unsigned lds_byte,
                                                  const _Float16* gptr) {
  asm volatile("global_load_async_to_lds_b128 %0, %1, off"
               :: "v"(lds_byte), "v"((unsigned long long)(size_t)gptr)
               : "memory");
}

// ---------------- kernel 0a: L2-normalize rows of x -> xn (f16) ----------------
__global__ void __launch_bounds__(256) k_norm(const float* __restrict__ x,
                                              const float* __restrict__ g,
                                              _Float16* __restrict__ xn) {
  __shared__ float red[256];
  int row = blockIdx.x;                        // b*N + n
  int d   = threadIdx.x;                       // 0..255 == D
  float v = x[(size_t)row * DD + d];
  red[d] = v * v;
  __syncthreads();
  for (int off = 128; off > 0; off >>= 1) {
    if (d < off) red[d] += red[d + off];
    __syncthreads();
  }
  float nrm = fmaxf(sqrtf(red[0]), 1e-5f);
  float scale = (1.0f / nrm) * 0.0625f * g[0]; // d^-0.5 = 1/16
  xn[(size_t)row * DD + d] = (_Float16)(v * scale);
}

// ---------------- kernel 0b: convert weights to f16 ----------------
__global__ void __launch_bounds__(256) k_cvt(const float* __restrict__ wuv,
                                             const float* __restrict__ wo,
                                             _Float16* __restrict__ wuv16,
                                             _Float16* __restrict__ wo16) {
  int i = blockIdx.x * 256 + threadIdx.x;
  if (i < NC * DD) wuv16[i] = (_Float16)wuv[i];
  if (i < DD * EE) wo16[i]  = (_Float16)wo[i];
}

// ---------------- kernel 1: uv = silu(xn @ Wuv^T); split into uT, vT, q, k ------
// grid: (NC/128, N/64, B), 256 threads = 8 waves (4 M x 2 C), 4 tiles/wave.
__global__ void __launch_bounds__(256) k_uv(const _Float16* __restrict__ xn,
                                            const _Float16* __restrict__ wuv16,
                                            const float* __restrict__ gamma,
                                            const float* __restrict__ beta,
                                            _Float16* __restrict__ uT,
                                            _Float16* __restrict__ vT,
                                            _Float16* __restrict__ q16,
                                            _Float16* __restrict__ k16) {
  int b  = blockIdx.z;
  int mb = blockIdx.y * 64;
  int cb = blockIdx.x * 128;
  int w  = threadIdx.x >> 5;
  int l  = threadIdx.x & 31;
  int wmi = w & 3, wcj = w >> 2;               // 4 x 2 wave grid
  int mrow = mb + wmi * 16;
  v8f acc[4] = {};
  const _Float16* abase = xn + ((size_t)(b * NN) + mrow) * DD;
  for (int kk = 0; kk < DD / 32; ++kk) {
    v16h a = load_frag_a(abase + kk * 32, DD);
    for (int t = 0; t < 4; ++t) {
      int col = cb + wcj * 64 + t * 16;
      v16h bf = load_frag_b(wuv16 + (size_t)col * DD + kk * 32, DD);
      acc[t] = wmma16(a, bf, acc[t]);
    }
  }
  int rbase = (l >> 4) * 8;
  int cl = l & 15;
  int m0 = mrow + rbase;
  for (int t = 0; t < 4; ++t) {
    int col = cb + wcj * 64 + t * 16 + cl;     // 0..1151
    float vals[8];
    for (int r = 0; r < 8; ++r) {
      float xv = acc[t][r];
      vals[r] = xv / (1.0f + __expf(-xv));     // silu
    }
    if (col < EE) {                            // u, stored transposed (B,E,N)
      v8h st;
      for (int r = 0; r < 8; ++r) st[r] = (_Float16)vals[r];
      *(v8h*)(uT + ((size_t)b * EE + col) * NN + m0) = st;
    } else if (col < 2 * EE) {                 // v, stored transposed (B,E,N)
      int e = col - EE;
      v8h st;
      for (int r = 0; r < 8; ++r) st[r] = (_Float16)vals[r];
      *(v8h*)(vT + ((size_t)b * EE + e) * NN + m0) = st;
    } else {                                   // base -> q,k  (B,N,S)
      int s = col - 2 * EE;
      float g0 = gamma[s], b0 = beta[s];
      float g1 = gamma[SS + s], b1 = beta[SS + s];
      for (int r = 0; r < 8; ++r) {
        size_t off = ((size_t)(b * NN) + m0 + r) * SS + s;
        q16[off] = (_Float16)(vals[r] * g0 + b0);
        k16[off] = (_Float16)(vals[r] * g1 + b1);
      }
    }
  }
}

// ---------------- kernel 2: squared-ReLU linear attention ----------------
// grid: (N/64, B), 512 threads = 16 waves. BLOCK_M=64, BLOCK_N=64.
// V key-blocks are staged to LDS with async copies (double-buffered): each
// 64KB block is fetched ONCE per workgroup instead of 4x redundantly, and the
// copy of block kb+1 overlaps the compute of block kb (s_wait_asynccnt 8).
__global__ void __launch_bounds__(512) k_attn(const _Float16* __restrict__ q16,
                                              const _Float16* __restrict__ k16,
                                              const _Float16* __restrict__ vT,
                                              const _Float16* __restrict__ uT,
                                              const float* __restrict__ w_rel,
                                              _Float16* __restrict__ attn) {
  __shared__ __align__(16) _Float16 P[64 * VLD];        //  9 KB P tile
  __shared__ __align__(16) _Float16 Vs[2][EE * VLD];    // 2x72 KB V tiles
  int b   = blockIdx.y;
  int qb  = blockIdx.x * 64;
  int tid = threadIdx.x;
  int w   = tid >> 5;
  int l   = tid & 31;
  int smi = w & 3, snj = w >> 2;                  // 4 x 4 score tile grid
  const float inv_s = 0.088388347648318447f;      // 1/sqrt(128)
  const int NKB = NN / 64;

  // Q fragments are loop invariant: keep them in VGPRs (4 x v16h).
  v16h qa[4];
  const _Float16* qbase = q16 + ((size_t)(b * NN) + qb + smi * 16) * SS;
  for (int kk = 0; kk < 4; ++kk) qa[kk] = load_frag_a(qbase + kk * 32, SS);

  v8f acc[8] = {};
  int rbase = (l >> 4) * 8;
  int cl = l & 15;

  // Each of the 512 threads stages one V row (e = tid): 64 halves = 8 x b128.
  const _Float16* vrow = vT + ((size_t)b * EE + tid) * NN;   // + kb*64 later
  unsigned vlds[2];
  vlds[0] = (unsigned)(size_t)(&Vs[0][tid * VLD]);
  vlds[1] = (unsigned)(size_t)(&Vs[1][tid * VLD]);

  // prefetch key block 0
  for (int j = 0; j < 8; ++j) async_b128(vlds[0] + j * 16, vrow + j * 8);

  for (int kb = 0; kb < NKB; ++kb) {
    // ---- scores: S_tile = Q . K^T (K-dim = S = 128 -> 4 wmma) ----
    v8f sc = {};
    const _Float16* kp = k16 + ((size_t)(b * NN) + kb * 64 + snj * 16) * SS;
    for (int kk = 0; kk < 4; ++kk) {
      v16h kf = load_frag_b(kp + kk * 32, SS);
      sc = wmma16(qa[kk], kf, sc);
    }
    __syncthreads();   // P(kb-1) consumed AND V buffer (kb+1)&1 no longer read
    // prefetch V block kb+1 into the other buffer (overlaps P-write below)
    if (kb + 1 < NKB) {
      const _Float16* vsrc = vrow + (size_t)(kb + 1) * 64;
      unsigned dst = vlds[(kb + 1) & 1];
      for (int j = 0; j < 8; ++j) async_b128(dst + j * 16, vsrc + j * 8);
    }
    int gm0 = qb + smi * 16 + rbase;
    int gn  = kb * 64 + snj * 16 + cl;
    for (int r = 0; r < 8; ++r) {
      int idx = gn - (gm0 + r) + (NN - 1);      // w_rel[j - i + N - 1]
      float t = (sc[r] + w_rel[idx]) * inv_s;
      t = fmaxf(t, 0.0f);
      P[(smi * 16 + rbase + r) * VLD + snj * 16 + cl] = (_Float16)(t * t);
    }
    // wait only for the OLDER 8 async copies (block kb); keep kb+1 in flight
    if (kb + 1 < NKB) asm volatile("s_wait_asynccnt 8" ::: "memory");
    else              asm volatile("s_wait_asynccnt 0" ::: "memory");
    __syncthreads();                            // V(kb) + P(kb) visible to all
    // ---- PV: acc += P . V  (K-dim = 64 -> 2 wmma per E tile) ----
    const _Float16* vbuf = Vs[kb & 1];
    for (int kk = 0; kk < 2; ++kk) {
      v16h pa = load_frag_a(P + (smi * 16) * VLD + kk * 32, VLD);
      for (int t = 0; t < 8; ++t) {
        int e = snj * 128 + t * 16;
        v16h vf = load_frag_b(vbuf + (size_t)e * VLD + kk * 32, VLD);
        acc[t] = wmma16(pa, vf, acc[t]);
      }
    }
  }

  // ---- epilogue: attn = u * acc, store (B,N,E) f16 ----
  int m0 = qb + smi * 16 + rbase;
  for (int t = 0; t < 8; ++t) {
    int e = snj * 128 + t * 16 + cl;
    v8h uv = *(const v8h*)(uT + ((size_t)b * EE + e) * NN + m0);
    for (int r = 0; r < 8; ++r) {
      float o = acc[t][r] * (float)uv[r];
      attn[((size_t)(b * NN) + m0 + r) * EE + e] = (_Float16)o;
    }
  }
}

// ---------------- kernel 3: out = x*res_scale + attn @ Wo^T ----------------
// grid: (D/128, N/64, B), 256 threads = 8 waves (4 M x 2 D), 4 tiles/wave.
__global__ void __launch_bounds__(256) k_out(const _Float16* __restrict__ attn,
                                             const _Float16* __restrict__ wo16,
                                             const float* __restrict__ x,
                                             const float* __restrict__ res_scale,
                                             float* __restrict__ out) {
  int b  = blockIdx.z;
  int mb = blockIdx.y * 64;
  int db = blockIdx.x * 128;
  int w  = threadIdx.x >> 5;
  int l  = threadIdx.x & 31;
  int wmi = w & 3, wdj = w >> 2;
  v8f acc[4] = {};
  const _Float16* abase = attn + ((size_t)(b * NN) + mb + wmi * 16) * EE;
  for (int kk = 0; kk < EE / 32; ++kk) {
    v16h a = load_frag_a(abase + kk * 32, EE);
    for (int t = 0; t < 4; ++t) {
      int col = db + wdj * 64 + t * 16;
      v16h bf = load_frag_b(wo16 + (size_t)col * EE + kk * 32, EE);
      acc[t] = wmma16(a, bf, acc[t]);
    }
  }
  int rbase = (l >> 4) * 8, cl = l & 15;
  int m0 = mb + wmi * 16 + rbase;
  for (int t = 0; t < 4; ++t) {
    int d = db + wdj * 64 + t * 16 + cl;
    float rs = res_scale[d];
    for (int r = 0; r < 8; ++r) {
      size_t off = ((size_t)(b * NN) + m0 + r) * DD + d;
      out[off] = x[off] * rs + acc[t][r];
    }
  }
}

extern "C" void kernel_launch(void* const* d_in, const int* in_sizes, int n_in,
                              void* d_out, int out_size, void* d_ws, size_t ws_size,
                              hipStream_t stream) {
  const float* x         = (const float*)d_in[0];
  const float* W_uv      = (const float*)d_in[1];
  const float* W_o       = (const float*)d_in[2];
  const float* gamma     = (const float*)d_in[3];
  const float* beta      = (const float*)d_in[4];
  const float* w_rel     = (const float*)d_in[5];
  const float* g         = (const float*)d_in[6];
  const float* res_scale = (const float*)d_in[7];
  float* out = (float*)d_out;

  char* ws = (char*)d_ws;
  size_t off = 0;
  auto carve = [&](size_t bytes) -> char* {
    char* p = ws + off;
    off += (bytes + 255) & ~(size_t)255;
    return p;
  };
  _Float16* xn    = (_Float16*)carve((size_t)BB * NN * DD * 2);
  _Float16* wuv16 = (_Float16*)carve((size_t)NC * DD * 2);
  _Float16* wo16  = (_Float16*)carve((size_t)DD * EE * 2);
  _Float16* uT    = (_Float16*)carve((size_t)BB * EE * NN * 2);
  _Float16* vT    = (_Float16*)carve((size_t)BB * EE * NN * 2);
  _Float16* q16   = (_Float16*)carve((size_t)BB * NN * SS * 2);
  _Float16* k16   = (_Float16*)carve((size_t)BB * NN * SS * 2);
  _Float16* attn  = (_Float16*)carve((size_t)BB * NN * EE * 2);
  (void)in_sizes; (void)n_in; (void)out_size; (void)ws_size;

  k_norm<<<dim3(BB * NN), dim3(256), 0, stream>>>(x, g, xn);
  k_cvt<<<dim3((NC * DD + 255) / 256), dim3(256), 0, stream>>>(W_uv, W_o, wuv16, wo16);
  k_uv<<<dim3(NC / 128, NN / 64, BB), dim3(256), 0, stream>>>(xn, wuv16, gamma, beta,
                                                              uT, vT, q16, k16);
  k_attn<<<dim3(NN / 64, BB), dim3(512), 0, stream>>>(q16, k16, vT, uT, w_rel, attn);
  k_out<<<dim3(DD / 128, NN / 64, BB), dim3(256), 0, stream>>>(attn, wo16, x,
                                                               res_scale, out);
}
